// CurricularFace_48206712930719
// MI455X (gfx1250) — matrix operand order repaired and checked by reading, not compiled, not run on hardware
//
#include <hip/hip_runtime.h>
#include <hip/hip_bf16.h>

// ---------------------------------------------------------------------------
// CurricularFace: N=512 (rows), D=512 (K), C=100000 (cols)
// out = concat( ct*64 [N*C], origin_cos*64 [N*C] )  (fp32)
// ---------------------------------------------------------------------------

typedef _Float16 f16;
typedef __attribute__((ext_vector_type(16))) _Float16 v16h;
typedef __attribute__((ext_vector_type(8)))  _Float16 v8h;
typedef __attribute__((ext_vector_type(8)))  float    v8f;

#define NN 512
#define DD 512
#define CC 100000
#define NTOT ((size_t)NN * (size_t)CC)

// margin constants (MARGIN = 0.5)
#define COS_M     0.8775825618903728f
#define SIN_M     0.479425538604203f
#define THRESH   (-0.8775825618903728f)
#define MM_CONST  0.2397127693021015f
#define SCALE_F   64.0f
#define ALPHA_F   0.1f

// ---------------------------------------------------------------------------
// K0: reciprocal column norms of kernel [D, C] (row-major)
// ---------------------------------------------------------------------------
__global__ __launch_bounds__(256) void k_colnorm(const float* __restrict__ ker,
                                                 float* __restrict__ rcn) {
    int c = blockIdx.x * 256 + threadIdx.x;
    if (c >= CC) return;
    float s = 0.0f;
    #pragma unroll 8
    for (int k = 0; k < DD; ++k) {
        float v = ker[(size_t)k * CC + c];
        s += v * v;
    }
    rcn[c] = rsqrtf(s);
}

// ---------------------------------------------------------------------------
// K1: row-normalize embeddings [N, D] -> f16 A, store 1/rownorm
// one block (256 threads) per row
// ---------------------------------------------------------------------------
__global__ __launch_bounds__(256) void k_normemb(const float* __restrict__ emb,
                                                 f16* __restrict__ A,
                                                 float* __restrict__ rrn) {
    __shared__ float red[256];
    int i = blockIdx.x, t = threadIdx.x;
    float x0 = emb[(size_t)i * DD + t];
    float x1 = emb[(size_t)i * DD + t + 256];
    red[t] = x0 * x0 + x1 * x1;
    __syncthreads();
    for (int s = 128; s > 0; s >>= 1) {
        if (t < s) red[t] += red[t + s];
        __syncthreads();
    }
    float r = rsqrtf(red[0]);
    A[(size_t)i * DD + t]       = (f16)(x0 * r);
    A[(size_t)i * DD + t + 256] = (f16)(x1 * r);
    if (t == 0) rrn[i] = r;
}

// ---------------------------------------------------------------------------
// K2: transpose + column-scale + f16 convert:  Bt[c*512 + k] = ker[k*C+c]*rcn[c]
// 32x32 tiles via LDS; block (32,8); grid (C/32, D/32) = (3125, 16)
// ---------------------------------------------------------------------------
__global__ __launch_bounds__(256) void k_transconv(const float* __restrict__ ker,
                                                   const float* __restrict__ rcn,
                                                   f16* __restrict__ Bt) {
    __shared__ float tile[32][33];
    int tx = threadIdx.x, ty = threadIdx.y;
    int c0 = blockIdx.x * 32, k0 = blockIdx.y * 32;
    #pragma unroll
    for (int j = 0; j < 4; ++j) {
        int kl = ty + j * 8;
        tile[kl][tx] = ker[(size_t)(k0 + kl) * CC + (c0 + tx)];
    }
    __syncthreads();
    #pragma unroll
    for (int j = 0; j < 4; ++j) {
        int cl = ty + j * 8;
        int c  = c0 + cl;
        float scale = rcn[c];
        Bt[(size_t)c * DD + (k0 + tx)] = (f16)(tile[tx][cl] * scale);
    }
}

// ---------------------------------------------------------------------------
// K3: per-row target logit (fp32), cos_theta_m, final_target_logit
// one block (256 threads) per row
// ---------------------------------------------------------------------------
__global__ __launch_bounds__(256) void k_target(const float* __restrict__ emb,
                                                const float* __restrict__ ker,
                                                const int* __restrict__ labels,
                                                const float* __restrict__ rrn,
                                                const float* __restrict__ rcn,
                                                float* __restrict__ tl,
                                                float* __restrict__ ctm,
                                                float* __restrict__ ftl) {
    __shared__ float red[256];
    int i = blockIdx.x, t = threadIdx.x;
    int lab = labels[i];
    float s = 0.0f;
    for (int k = t; k < DD; k += 256)
        s += emb[(size_t)i * DD + k] * ker[(size_t)k * CC + lab];
    red[t] = s;
    __syncthreads();
    for (int st = 128; st > 0; st >>= 1) {
        if (t < st) red[t] += red[t + st];
        __syncthreads();
    }
    if (t == 0) {
        float cosv = red[0] * rrn[i] * rcn[lab];
        cosv = fminf(1.0f, fmaxf(-1.0f, cosv));
        float sinv = sqrtf(fmaxf(0.0f, 1.0f - cosv * cosv));
        float cm = cosv * COS_M - sinv * SIN_M;
        tl[i]  = cosv;
        ctm[i] = cm;
        ftl[i] = (cosv > THRESH) ? cm : (cosv - MM_CONST);
    }
}

// ---------------------------------------------------------------------------
// K4: t_new = mean(target_logit)*ALPHA + (1-ALPHA)*t   (single block, 512 thr)
// ---------------------------------------------------------------------------
__global__ __launch_bounds__(512) void k_tnew(const float* __restrict__ tl,
                                              const float* __restrict__ t_in,
                                              float* __restrict__ tnew) {
    __shared__ float red[512];
    int t = threadIdx.x;
    red[t] = tl[t];
    __syncthreads();
    for (int s = 256; s > 0; s >>= 1) {
        if (t < s) red[t] += red[t + s];
        __syncthreads();
    }
    if (t == 0)
        tnew[0] = (red[0] / (float)NN) * ALPHA_F + (1.0f - ALPHA_F) * t_in[0];
}

// ---------------------------------------------------------------------------
// K5: WMMA GEMM + CurricularFace epilogue
// wave32: each wave computes a 16(M) x 32(N) tile; 8 waves per block.
// A f16 row-major [512,512]; Bt f16 column-major [C rows of 512 K each].
// A-frag and B-frag both hold contiguous k = 16h + j (same order in A and B).
// ---------------------------------------------------------------------------
__global__ __launch_bounds__(256) void k_gemm(const f16* __restrict__ A,
                                              const f16* __restrict__ Bt,
                                              const float* __restrict__ ctm,
                                              const float* __restrict__ ftl,
                                              const float* __restrict__ tnew,
                                              const int* __restrict__ labels,
                                              float* __restrict__ out) {
    const int lane  = threadIdx.x & 31;
    const int wvb   = threadIdx.x >> 5;             // wave in block, 0..7
    const int wid   = blockIdx.x * 8 + wvb;         // global wave id
    const int mtile = wid & 31;                     // 0..31  (512/16)
    const int ntile = wid >> 5;                     // 0..3124 (100000/32)
    const int mbase = mtile * 16;
    const int nbase = ntile * 32;
    const int half  = lane >> 4;                    // 0: lanes 0-15, 1: 16-31
    const int l16   = lane & 15;

    // A: row = mbase + l16; halves[j] = K(16h + j)
    const f16* arow  = A + (size_t)(mbase + l16) * DD;
    // B: col = nbase + (tile*16) + l16; halves[j] = K(16h + j)
    const f16* bcol0 = Bt + (size_t)(nbase + l16) * DD + 16 * half;
    const f16* bcol1 = bcol0 + (size_t)16 * DD;

    v8f acc0 = {};
    v8f acc1 = {};

    #pragma unroll 4
    for (int k = 0; k < DD; k += 32) {
        v16h a = *(const v16h*)(arow + k + 16 * half);
        v16h b0 = *(const v16h*)(bcol0 + k);
        v16h b1 = *(const v16h*)(bcol1 + k);
        acc0 = __builtin_amdgcn_wmma_f32_16x16x32_f16(false, a, false, b0,
                                                      (short)0, acc0, false, false);
        acc1 = __builtin_amdgcn_wmma_f32_16x16x32_f16(false, a, false, b1,
                                                      (short)0, acc1, false, false);
    }

    // Epilogue: D layout -> VGPR v, lanes 0-15: (M=v, N=l16); lanes 16-31: (M=v+8, N=l16)
    const float tnv = tnew[0];
    const int col0 = nbase + l16;
    const int col1 = nbase + 16 + l16;
    #pragma unroll
    for (int v = 0; v < 8; ++v) {
        int row = mbase + v + 8 * half;
        float cmr = ctm[row];
        float ftr = ftl[row];
        int   lab = labels[row];
        size_t base = (size_t)row * CC;

        float c0 = fminf(1.0f, fmaxf(-1.0f, acc0[v]));
        float ct0 = (c0 > cmr) ? c0 * (tnv + c0) : c0;
        if (col0 == lab) ct0 = ftr;
        out[base + col0]        = ct0 * SCALE_F;
        out[NTOT + base + col0] = c0 * SCALE_F;

        float c1 = fminf(1.0f, fmaxf(-1.0f, acc1[v]));
        float ct1 = (c1 > cmr) ? c1 * (tnv + c1) : c1;
        if (col1 == lab) ct1 = ftr;
        out[base + col1]        = ct1 * SCALE_F;
        out[NTOT + base + col1] = c1 * SCALE_F;
    }
}

// ---------------------------------------------------------------------------
// Launcher
// ---------------------------------------------------------------------------
extern "C" void kernel_launch(void* const* d_in, const int* in_sizes, int n_in,
                              void* d_out, int out_size, void* d_ws, size_t ws_size,
                              hipStream_t stream) {
    const float* emb    = (const float*)d_in[0];   // [512, 512]
    const float* ker    = (const float*)d_in[1];   // [512, 100000]
    const float* t_in   = (const float*)d_in[2];   // [1]
    const int*   labels = (const int*)d_in[3];     // [512]
    float* out = (float*)d_out;

    char* ws = (char*)d_ws;
    size_t off = 0;
    f16*   Af16 = (f16*)(ws + off);   off += (size_t)NN * DD * sizeof(f16);    // 512 KB
    f16*   Bt   = (f16*)(ws + off);   off += (size_t)CC * DD * sizeof(f16);    // 102.4 MB
    float* rcn  = (float*)(ws + off); off += (size_t)CC * sizeof(float);       // 400 KB
    off = (off + 255) & ~(size_t)255;
    float* rrn  = (float*)(ws + off); off += NN * sizeof(float);
    float* tl   = (float*)(ws + off); off += NN * sizeof(float);
    float* ctm  = (float*)(ws + off); off += NN * sizeof(float);
    float* ftl  = (float*)(ws + off); off += NN * sizeof(float);
    float* tnew = (float*)(ws + off); off += 256;

    // 1) reciprocal column norms of kernel
    k_colnorm<<<(CC + 255) / 256, 256, 0, stream>>>(ker, rcn);
    // 2) row-normalize embeddings -> f16 A
    k_normemb<<<NN, 256, 0, stream>>>(emb, Af16, rrn);
    // 3) transpose + scale + convert kernel -> Bt (f16, column-major)
    k_transconv<<<dim3(CC / 32, DD / 32), dim3(32, 8), 0, stream>>>(ker, rcn, Bt);
    // 4) per-row target logit + margin terms
    k_target<<<NN, 256, 0, stream>>>(emb, ker, labels, rrn, rcn, tl, ctm, ftl);
    // 5) EMA scalar t_new
    k_tnew<<<1, 512, 0, stream>>>(tl, t_in, tnew);
    // 6) WMMA GEMM + epilogue: 32 M-tiles * 3125 N-tiles = 100000 waves = 12500 blocks
    k_gemm<<<(32 * (CC / 32)) / 8, 256, 0, stream>>>(Af16, Bt, ctm, ftl, tnew, labels, out);
}
